// RelPartialLearnableMultiHeadAttn_88613765251217
// MI455X (gfx1250) — matrix operand order repaired and checked by reading, not compiled, not run on hardware
//
#include <hip/hip_runtime.h>

// ---------------- problem constants ----------------
#define NHEAD  16
#define DMODEL 1024
#define DHEAD  64
#define QLEN   1024
#define MLEN   1024
#define KLEN   2048   // QLEN + MLEN
#define RLEN   2048
#define BSZ    4
#define SCALE  0.125f // 1/sqrt(64)
#define LN_EPS 1e-5f

// ---------------- types ----------------
typedef __bf16 bf16;
typedef __attribute__((ext_vector_type(16))) __bf16 v16bf;
typedef __attribute__((ext_vector_type(8)))  float  v8f;
typedef __attribute__((ext_vector_type(4)))  unsigned int u32x4;
typedef __attribute__((ext_vector_type(4)))  float  f32x4;

union FragAB { v16bf v; u32x4 q[2]; };   // 32 bytes: one WMMA A or B operand
union FragC  { v8f   v; float f[8]; };   // 32 bytes: one WMMA C/D accumulator

__device__ __forceinline__ unsigned short f2bfu(float f) {
  unsigned int u = __builtin_bit_cast(unsigned int, f);
  unsigned int r = u + 0x7FFFu + ((u >> 16) & 1u);   // round-to-nearest-even
  return (unsigned short)(r >> 16);
}
__device__ __forceinline__ bf16 f2bf(float f) {
  return __builtin_bit_cast(bf16, f2bfu(f));
}

// CDNA5 async global->LDS copy, 16B per lane. ldsOff = LDS byte offset
// (low 32 bits of flat shared address), g = 64-bit global address.
__device__ __forceinline__ void async_g2l_b128(unsigned ldsOff, const void* g) {
  asm volatile("global_load_async_to_lds_b128 %0, %1, off"
               :: "v"(ldsOff), "v"((unsigned long long)(uintptr_t)g)
               : "memory");
}

// ---------------- fp32 -> bf16 convert (vectorized x4) ----------------
__global__ __launch_bounds__(256)
void k_cvt_bf16(const float* __restrict__ src, bf16* __restrict__ dst, long n4) {
  long i = (long)blockIdx.x * 256 + threadIdx.x;
  if (i >= n4) return;
  f32x4 f = ((const f32x4*)src)[i];
  unsigned long long p =
      (unsigned long long)f2bfu(f.x)
    | ((unsigned long long)f2bfu(f.y) << 16)
    | ((unsigned long long)f2bfu(f.z) << 32)
    | ((unsigned long long)f2bfu(f.w) << 48);
  ((unsigned long long*)dst)[i] = p;
}

// ---------------- head packing kernels ----------------
// Qh [i*BSZ+b][n*64+d] -> qw/qr [b][n][i][d] (+ biases), bf16
__global__ __launch_bounds__(256)
void k_pack_q(const float* __restrict__ Qh, const float* __restrict__ rwb,
              const float* __restrict__ rrb, bf16* __restrict__ qw,
              bf16* __restrict__ qr) {
  long idx = (long)blockIdx.x * 256 + threadIdx.x;         // over 4096*1024
  int col = (int)(idx & (DMODEL - 1));
  long row = idx >> 10;
  int i = (int)(row >> 2), b = (int)(row & 3);
  int n = col >> 6, d = col & 63;
  float q = Qh[idx];
  long o = ((((long)b * NHEAD + n) * QLEN + i) * DHEAD + d);
  qw[o] = f2bf(q + rwb[col]);
  qr[o] = f2bf(q + rrb[col]);
}

// Kh [j*BSZ+b][n*64+d] -> khT [b][n][d][j], bf16 (transposed for B-fragments)
__global__ __launch_bounds__(256)
void k_pack_k(const float* __restrict__ Kh, bf16* __restrict__ khT) {
  long idx = (long)blockIdx.x * 256 + threadIdx.x;         // over 8192*1024
  int col = (int)(idx & (DMODEL - 1));
  long row = idx >> 10;
  int j = (int)(row >> 2), b = (int)(row & 3);
  int n = col >> 6, d = col & 63;
  khT[((((long)b * NHEAD + n) * DHEAD + d) * KLEN + j)] = f2bf(Kh[idx]);
}

// Vh [j*BSZ+b][n*64+d] -> vh [b][n][j][d], bf16
__global__ __launch_bounds__(256)
void k_pack_v(const float* __restrict__ Vh, bf16* __restrict__ vh) {
  long idx = (long)blockIdx.x * 256 + threadIdx.x;
  int col = (int)(idx & (DMODEL - 1));
  long row = idx >> 10;
  int j = (int)(row >> 2), b = (int)(row & 3);
  int n = col >> 6, d = col & 63;
  vh[((((long)b * NHEAD + n) * KLEN + j) * DHEAD + d)] = f2bf(Vh[idx]);
}

// Rh [m][n*64+d] -> rhT [n][d][m], bf16
__global__ __launch_bounds__(256)
void k_pack_r(const float* __restrict__ Rh, bf16* __restrict__ rhT) {
  long idx = (long)blockIdx.x * 256 + threadIdx.x;         // over 2048*1024
  int col = (int)(idx & (DMODEL - 1));
  long m = idx >> 10;
  int n = col >> 6, d = col & 63;
  rhT[(((long)n * DHEAD + d) * RLEN + m)] = f2bf(Rh[idx]);
}

// ---------------- generic batched bf16 WMMA GEMM ----------------
// C[M,N] (f32) = A[M,K] (bf16, row major) @ B[K,N] (bf16, row major)
// block tile 128x128, 8 waves (2 M x 4 N), wave tile 64x32, K-step 32.
// Double-buffered LDS fed by async global->LDS copies, software-pipelined:
// tile s+1 streams in (ASYNCcnt) while tile s feeds the WMMAs.
__global__ __launch_bounds__(256)
void k_gemm_bf16(const bf16* __restrict__ A, long lda, long sA,
                 const bf16* __restrict__ B, long ldb, long sB,
                 float* __restrict__ C, long ldc, long sC,
                 int M, int N, int K) {
  __shared__ bf16 As[2][128][40];    // 32 K + pad
  __shared__ bf16 Bs[2][32][136];    // 128 N + pad
  const int tid = threadIdx.x;
  const int bm = blockIdx.y, bn = blockIdx.x, bz = blockIdx.z;
  A += (long)bz * sA; B += (long)bz * sB; C += (long)bz * sC;
  const int wid = tid >> 5, lane = tid & 31;
  const int wm = wid >> 2, wn = wid & 3;        // 2 x 4 wave grid
  const int hf = lane >> 4, l15 = lane & 15;

  // per-thread async copy slots: 2 x 16B chunks of A, 2 x 16B chunks of B
  const int arow = tid >> 2, acol = (tid & 3) * 8;       // A: rows arow, arow+64
  const int brow = tid >> 4, bcol = (tid & 15) * 8;      // B: rows brow, brow+16
  const bf16* gA0 = A + (long)(bm * 128 + arow) * lda + acol;
  const bf16* gA1 = gA0 + (long)64 * lda;
  const bf16* gB0 = B + (long)brow * ldb + bn * 128 + bcol;
  const bf16* gB1 = gB0 + (long)16 * ldb;
  const unsigned lAsBase = (unsigned)(uintptr_t)&As[0][0][0];
  const unsigned lBsBase = (unsigned)(uintptr_t)&Bs[0][0][0];
  const unsigned lA0 = lAsBase + (unsigned)(arow * 40 + acol) * 2;
  const unsigned lA1 = lA0 + 64u * 40u * 2u;
  const unsigned lB0 = lBsBase + (unsigned)(brow * 136 + bcol) * 2;
  const unsigned lB1 = lB0 + 16u * 136u * 2u;
  const unsigned strideAbuf = 128u * 40u * 2u;   // bytes per As buffer
  const unsigned strideBbuf = 32u * 136u * 2u;   // bytes per Bs buffer

  auto issue_tile = [&](int k0, int buf) {
    const unsigned oa = (unsigned)buf * strideAbuf;
    const unsigned ob = (unsigned)buf * strideBbuf;
    async_g2l_b128(lA0 + oa, gA0 + k0);
    async_g2l_b128(lA1 + oa, gA1 + k0);
    async_g2l_b128(lB0 + ob, gB0 + (long)k0 * ldb);
    async_g2l_b128(lB1 + ob, gB1 + (long)k0 * ldb);
  };

  FragC acc[4][2] = {};
  const int nsteps = K / 32;

  issue_tile(0, 0);
  for (int s = 0; s < nsteps; ++s) {
    const int buf = s & 1;
    if (s + 1 < nsteps) {
      issue_tile((s + 1) * 32, buf ^ 1);
      asm volatile("s_wait_asynccnt 0x4" ::: "memory");  // tile s landed
    } else {
      asm volatile("s_wait_asynccnt 0x0" ::: "memory");
    }
    __syncthreads();

    FragAB af[4], bfg[2];
    // A fragment: lane -> M = l15, halves split K {0..7,16..23} / {8..15,24..31}
    for (int mt = 0; mt < 4; ++mt) {
      int mrow = wm * 64 + mt * 16 + l15;
      af[mt].q[0] = *(const u32x4*)&As[buf][mrow][hf * 8];
      af[mt].q[1] = *(const u32x4*)&As[buf][mrow][16 + hf * 8];
    }
    // B fragment: lane -> N = l15, elements = K in same split as A
    for (int nt = 0; nt < 2; ++nt) {
      int nb = wn * 32 + nt * 16;
      for (int e = 0; e < 16; ++e)
        bfg[nt].v[e] = Bs[buf][(e < 8 ? hf * 8 + e : 16 + hf * 8 + e - 8)][nb + l15];
    }
    for (int mt = 0; mt < 4; ++mt)
      for (int nt = 0; nt < 2; ++nt)
        acc[mt][nt].v = __builtin_amdgcn_wmma_f32_16x16x32_bf16(
            false, af[mt].v, false, bfg[nt].v, (short)0, acc[mt][nt].v,
            false, false);
    __syncthreads();   // all waves done with buf before it is refilled
  }

  // C/D layout: lane l, vgpr v -> M = v + 8*(l>>4), N = l&15
  for (int mt = 0; mt < 4; ++mt)
    for (int nt = 0; nt < 2; ++nt) {
      int r0 = bm * 128 + wm * 64 + mt * 16 + hf * 8;
      int c0 = bn * 128 + wn * 32 + nt * 16 + l15;
      for (int v = 0; v < 8; ++v)
        C[(long)(r0 + v) * ldc + c0] = acc[mt][nt].f[v];
    }
}

// ---------------- fused attention (flash-style, per batch) ----------------
// grid (QLEN/64, NHEAD), block 128 (4 waves). Wave w owns 16 q rows.
// shifted BD[i,j] = BDraw[i, j + QLEN-1 - i]  (valid wherever unmasked)
__global__ __launch_bounds__(128)
void k_attn(int b, const bf16* __restrict__ qw, const bf16* __restrict__ qr,
            const bf16* __restrict__ khT, const bf16* __restrict__ vh,
            const float* __restrict__ BD, bf16* __restrict__ av) {
  __shared__ bf16 Plds[4][16][32];   // per-wave P transpose patch
  const int lane = threadIdx.x & 31, w = threadIdx.x >> 5;
  const int n = blockIdx.y;
  const int qi0 = blockIdx.x * 64 + w * 16;
  const int hf = lane >> 4, l15 = lane & 15;

  const bf16* qwb = qw + ((((long)b * NHEAD + n) * QLEN + qi0) * DHEAD);
  const bf16* qrb = qr + ((((long)b * NHEAD + n) * QLEN + qi0) * DHEAD);
  const bf16* kb  = khT + (((long)b * NHEAD + n) * DHEAD) * (long)KLEN; // [d][klen]
  const bf16* vb  = vh  + (((long)b * NHEAD + n) * KLEN) * (long)DHEAD; // [j][d]
  const float* bdb = BD + (long)n * QLEN * RLEN;                        // [i][m]

  // q fragments (A operands), 2 K-chunks of 32 each, resident for whole loop
  FragAB aw[2], ar[2];
  {
    const bf16* rw = qwb + (long)l15 * DHEAD;
    const bf16* rr = qrb + (long)l15 * DHEAD;
    for (int c = 0; c < 2; ++c) {
      aw[c].q[0] = *(const u32x4*)(rw + c * 32 + hf * 8);
      aw[c].q[1] = *(const u32x4*)(rw + c * 32 + 16 + hf * 8);
      ar[c].q[0] = *(const u32x4*)(rr + c * 32 + hf * 8);
      ar[c].q[1] = *(const u32x4*)(rr + c * 32 + 16 + hf * 8);
    }
  }

  FragC O[4] = {};
  float mst[8], lst[8];
  for (int v = 0; v < 8; ++v) { mst[v] = -3.0e38f; lst[v] = 0.f; }

  const int jmax = min(KLEN, qi0 + 15 + MLEN + 1);
  for (int j0 = 0; j0 < jmax; j0 += 32) {
    FragC s[2] = {};
    // AC via WMMA: S[16,16] += Q[16,64] * K^T (two 16-wide j subtiles)
    for (int jt = 0; jt < 2; ++jt)
      for (int c = 0; c < 2; ++c) {
        FragAB kf;
        const bf16* kp = kb + (long)(c * 32) * KLEN + j0 + jt * 16 + l15;
        for (int e = 0; e < 16; ++e)
          kf.v[e] = kp[(long)(e < 8 ? hf * 8 + e : 16 + hf * 8 + e - 8) * KLEN];
        s[jt].v = __builtin_amdgcn_wmma_f32_16x16x32_bf16(
            false, aw[c].v, false, kf.v, (short)0, s[jt].v, false, false);
      }
    // + shifted BD (q.r term via the precomputed BDraw GEMM), scale, mask
    for (int v = 0; v < 8; ++v) {
      int i = qi0 + hf * 8 + v;
      for (int jt = 0; jt < 2; ++jt) {
        int j = j0 + jt * 16 + l15;
        int m = j + (QLEN - 1) - i;
        float sv = s[jt].f[v];
        float bd = (m < RLEN) ? bdb[(long)i * RLEN + m] : 0.f;
        sv = (sv + bd) * SCALE;
        if (j > i + MLEN) sv = -1e30f;   // causal mask with memory offset
        s[jt].f[v] = sv;
      }
      // online softmax: row max/sum reduce across 16 lanes of this half
      float rm = fmaxf(s[0].f[v], s[1].f[v]);
      for (int o = 1; o < 16; o <<= 1) rm = fmaxf(rm, __shfl_xor(rm, o, 32));
      float nm = fmaxf(mst[v], rm);
      float p0 = __expf(s[0].f[v] - nm);
      float p1 = __expf(s[1].f[v] - nm);
      s[0].f[v] = p0; s[1].f[v] = p1;
      float rs = p0 + p1;
      for (int o = 1; o < 16; o <<= 1) rs += __shfl_xor(rs, o, 32);
      float alpha = __expf(mst[v] - nm);
      lst[v] = lst[v] * alpha + rs;
      mst[v] = nm;
      for (int t = 0; t < 4; ++t) O[t].f[v] *= alpha;
    }
    // transpose P through per-wave LDS into an A fragment (16x32 bf16)
    for (int v = 0; v < 8; ++v) {
      int row = hf * 8 + v;
      Plds[w][row][l15]      = f2bf(s[0].f[v]);
      Plds[w][row][16 + l15] = f2bf(s[1].f[v]);
    }
    asm volatile("s_wait_dscnt 0x0" ::: "memory");  // wave-local LDS RAW
    FragAB pf;
    {
      const bf16* pp = &Plds[w][l15][0];
      pf.q[0] = *(const u32x4*)(pp + hf * 8);
      pf.q[1] = *(const u32x4*)(pp + 16 + hf * 8);
    }
    // O[16,64] += P[16,32] @ V[32,64]
    for (int t = 0; t < 4; ++t) {
      FragAB vf;
      const bf16* vp = vb + (long)j0 * DHEAD + t * 16 + l15;
      for (int e = 0; e < 16; ++e)
        vf.v[e] = vp[(long)(e < 8 ? hf * 8 + e : 16 + hf * 8 + e - 8) * DHEAD];
      O[t].v = __builtin_amdgcn_wmma_f32_16x16x32_bf16(
          false, pf.v, false, vf.v, (short)0, O[t].v, false, false);
    }
    asm volatile("s_wait_dscnt 0x0" ::: "memory");  // before next-iter overwrite
  }

  // normalize and store attn_vec as bf16 [i][b][n*64+d]
  for (int v = 0; v < 8; ++v) {
    float inv = 1.f / lst[v];
    int i = qi0 + hf * 8 + v;
    bf16* op = av + (((long)i * BSZ + b) * DMODEL + n * DHEAD);
    for (int t = 0; t < 4; ++t)
      op[t * 16 + l15] = f2bf(O[t].f[v] * inv);
  }
}

// ---------------- residual + layernorm ----------------
__device__ __forceinline__ float wave_sum32(float s) {
  for (int o = 1; o < 32; o <<= 1) s += __shfl_xor(s, o, 32);
  return s;
}
__global__ __launch_bounds__(256)
void k_ln(const float* __restrict__ x, const float* __restrict__ ao,
          const float* __restrict__ gamma, const float* __restrict__ beta,
          float* __restrict__ out) {
  __shared__ float red[8];
  const long row = blockIdx.x;
  const int tid = threadIdx.x, wid = tid >> 5, lane = tid & 31;
  const float* xr = x + row * DMODEL;
  const float* ar = ao + row * DMODEL;
  float v[4]; float s = 0.f;
  for (int e = 0; e < 4; ++e) {
    int c = tid * 4 + e;
    v[e] = xr[c] + ar[c];
    s += v[e];
  }
  s = wave_sum32(s);
  if (lane == 0) red[wid] = s;
  __syncthreads();
  float t0 = (tid < 8) ? red[tid] : 0.f;
  __syncthreads();
  if (wid == 0) {
    t0 = wave_sum32(t0);
    if (lane == 0) red[0] = t0;
  }
  __syncthreads();
  const float mu = red[0] * (1.f / DMODEL);
  __syncthreads();
  float sv = 0.f;
  for (int e = 0; e < 4; ++e) { float d = v[e] - mu; sv += d * d; }
  sv = wave_sum32(sv);
  if (lane == 0) red[wid] = sv;
  __syncthreads();
  float t1 = (tid < 8) ? red[tid] : 0.f;
  __syncthreads();
  if (wid == 0) {
    t1 = wave_sum32(t1);
    if (lane == 0) red[0] = t1;
  }
  __syncthreads();
  const float var = red[0] * (1.f / DMODEL);
  const float rstd = rsqrtf(var + LN_EPS);
  for (int e = 0; e < 4; ++e) {
    int c = tid * 4 + e;
    out[row * DMODEL + c] = (v[e] - mu) * rstd * gamma[c] + beta[c];
  }
}

// ---------------- host launch ----------------
static inline int blks(long n, int t) { return (int)((n + t - 1) / t); }

extern "C" void kernel_launch(void* const* d_in, const int* in_sizes, int n_in,
                              void* d_out, int out_size, void* d_ws, size_t ws_size,
                              hipStream_t stream) {
  const float* query   = (const float*)d_in[0];
  const float* content = (const float*)d_in[1];
  const float* r       = (const float*)d_in[2];
  const float* mems    = (const float*)d_in[3];
  /* d_in[4] = attn_mask (recomputed analytically) */
  const float* Wq  = (const float*)d_in[5];
  const float* Wk  = (const float*)d_in[6];
  const float* Wv  = (const float*)d_in[7];
  const float* Wr  = (const float*)d_in[8];
  const float* Wo  = (const float*)d_in[9];
  const float* rwb = (const float*)d_in[10];
  const float* rrb = (const float*)d_in[11];
  const float* gam = (const float*)d_in[12];
  const float* bet = (const float*)d_in[13];
  float* out = (float*)d_out;

  char* ws = (char*)d_ws;
  size_t off = 0;
  auto alloc = [&](size_t bytes) -> void* {
    off = (off + 255) & ~(size_t)255;
    void* p = ws + off;
    off += bytes;
    return p;
  };

  const long nQ = (long)QLEN * BSZ * DMODEL;   // 4096*1024
  const long nC = (long)KLEN * BSZ * DMODEL;   // 8192*1024
  const long nW = (long)DMODEL * DMODEL;
  const long nR = (long)RLEN * DMODEL;
  const long nHq = (long)BSZ * NHEAD * QLEN * DHEAD;
  const long nHk = (long)BSZ * NHEAD * KLEN * DHEAD;
  const long nHr = (long)NHEAD * RLEN * DHEAD;

  bf16* Xq   = (bf16*)alloc(nQ * 2);
  bf16* Xcat = (bf16*)alloc(nC * 2);
  bf16* Wqb  = (bf16*)alloc(nW * 2);
  bf16* Wkb  = (bf16*)alloc(nW * 2);
  bf16* Wvb  = (bf16*)alloc(nW * 2);
  bf16* Wrb  = (bf16*)alloc(nW * 2);
  bf16* Wob  = (bf16*)alloc(nW * 2);
  bf16* Rb   = (bf16*)alloc(nR * 2);
  float* Qh  = (float*)alloc(nQ * 4);
  float* Kh  = (float*)alloc(nC * 4);
  float* Vh  = (float*)alloc(nC * 4);
  float* Rh  = (float*)alloc(nR * 4);
  bf16* qwh  = (bf16*)alloc(nHq * 2);
  bf16* qrh  = (bf16*)alloc(nHq * 2);
  bf16* khT  = (bf16*)alloc(nHk * 2);
  bf16* vhh  = (bf16*)alloc(nHk * 2);
  bf16* rhT  = (bf16*)alloc(nHr * 2);
  float* BDb = (float*)alloc((long)NHEAD * QLEN * RLEN * 4);  // per-batch reuse
  bf16* avb  = (bf16*)alloc(nQ * 2);
  float* aob = (float*)alloc(nQ * 4);

  // 1) fp32 -> bf16
  k_cvt_bf16<<<blks(nQ / 4, 256), 256, 0, stream>>>(query, Xq, nQ / 4);
  k_cvt_bf16<<<blks(nQ / 4, 256), 256, 0, stream>>>(mems, Xcat, nQ / 4); // mems = first MLEN*BSZ rows
  k_cvt_bf16<<<blks(nQ / 4, 256), 256, 0, stream>>>(content, Xcat + nQ, nQ / 4);
  k_cvt_bf16<<<blks(nW / 4, 256), 256, 0, stream>>>(Wq, Wqb, nW / 4);
  k_cvt_bf16<<<blks(nW / 4, 256), 256, 0, stream>>>(Wk, Wkb, nW / 4);
  k_cvt_bf16<<<blks(nW / 4, 256), 256, 0, stream>>>(Wv, Wvb, nW / 4);
  k_cvt_bf16<<<blks(nW / 4, 256), 256, 0, stream>>>(Wr, Wrb, nW / 4);
  k_cvt_bf16<<<blks(nW / 4, 256), 256, 0, stream>>>(Wo, Wob, nW / 4);
  k_cvt_bf16<<<blks(nR / 4, 256), 256, 0, stream>>>(r, Rb, nR / 4);

  // 2) projections (bf16 WMMA GEMMs, f32 out)
  k_gemm_bf16<<<dim3(DMODEL / 128, (QLEN * BSZ) / 128, 1), 256, 0, stream>>>(
      Xq, DMODEL, 0, Wqb, DMODEL, 0, Qh, DMODEL, 0, QLEN * BSZ, DMODEL, DMODEL);
  k_gemm_bf16<<<dim3(DMODEL / 128, (KLEN * BSZ) / 128, 1), 256, 0, stream>>>(
      Xcat, DMODEL, 0, Wkb, DMODEL, 0, Kh, DMODEL, 0, KLEN * BSZ, DMODEL, DMODEL);
  k_gemm_bf16<<<dim3(DMODEL / 128, (KLEN * BSZ) / 128, 1), 256, 0, stream>>>(
      Xcat, DMODEL, 0, Wvb, DMODEL, 0, Vh, DMODEL, 0, KLEN * BSZ, DMODEL, DMODEL);
  k_gemm_bf16<<<dim3(DMODEL / 128, RLEN / 128, 1), 256, 0, stream>>>(
      Rb, DMODEL, 0, Wrb, DMODEL, 0, Rh, DMODEL, 0, RLEN, DMODEL, DMODEL);

  // 3) head packing (+ biases, transposes)
  k_pack_q<<<blks(nQ, 256), 256, 0, stream>>>(Qh, rwb, rrb, qwh, qrh);
  k_pack_k<<<blks(nC, 256), 256, 0, stream>>>(Kh, khT);
  k_pack_v<<<blks(nC, 256), 256, 0, stream>>>(Vh, vhh);
  k_pack_r<<<blks(nR, 256), 256, 0, stream>>>(Rh, rhT);

  // 4) per batch: BD_raw GEMM (batched over heads) then fused attention
  for (int b = 0; b < BSZ; ++b) {
    const bf16* Aqr = qrh + (long)b * NHEAD * QLEN * DHEAD;
    k_gemm_bf16<<<dim3(RLEN / 128, QLEN / 128, NHEAD), 256, 0, stream>>>(
        Aqr, DHEAD, (long)QLEN * DHEAD,
        rhT, RLEN, (long)DHEAD * RLEN,
        BDb, RLEN, (long)QLEN * RLEN,
        QLEN, RLEN, DHEAD);
    k_attn<<<dim3(QLEN / 64, NHEAD), 128, 0, stream>>>(
        b, qwh, qrh, khT, vhh, BDb, avb);
  }

  // 5) output projection + residual layernorm
  k_gemm_bf16<<<dim3(DMODEL / 128, (QLEN * BSZ) / 128, 1), 256, 0, stream>>>(
      avb, DMODEL, 0, Wob, DMODEL, 0, aob, DMODEL, 0, QLEN * BSZ, DMODEL, DMODEL);
  k_ln<<<QLEN * BSZ, 256, 0, stream>>>(query, aob, gam, bet, out);
}
